// se_equivariant_net_73323681677596
// MI455X (gfx1250) — compile-verified, matrix operand order checked by reading, not run on hardware
//
#include <hip/hip_runtime.h>
#include <hip/hip_bf16.h>
#include <math.h>

#define NN 512
#define HH 4
#define DD 32
#define UU 64
#define LL 3
#define HC 48      // h_cat dim = D + H*H
#define KE 128     // padded phi_e input K (100 -> 128)
#define WPL 21504  // packed f16 weight halves per layer

typedef _Float16 f16;
typedef __attribute__((ext_vector_type(16))) _Float16 v16h;
typedef __attribute__((ext_vector_type(8)))  _Float16 h8;
typedef __attribute__((ext_vector_type(4)))  _Float16 h4;
typedef __attribute__((ext_vector_type(8)))  float    v8f;

__device__ __forceinline__ void lds_fence() {
    asm volatile("s_wait_dscnt 0" ::: "memory");
}

__device__ __forceinline__ float silu_f(float v) {
    return v / (1.f + __expf(-v));
}

// Load one 16-half WMMA operand fragment (A or B) given the documented
// 16-bit 16x32 layout: lanes<16 hold K {0..7,16..23}, lanes>=16 hold K {8..15,24..31}.
// p0 points at the first 8 contiguous halves for this lane.
__device__ __forceinline__ v16h ld_frag(const f16* p0) {
    h8 lo = *(const h8*)p0;
    h8 up = *(const h8*)(p0 + 16);
    v16h r;
#pragma unroll
    for (int t = 0; t < 8; ++t) { r[t] = lo[t]; r[t + 8] = up[t]; }
    return r;
}

// ---------------------------------------------------------------------------
// Weight prep: fp32 -> f16, transposed to [out_col][K] so B fragments are
// two contiguous 16-byte loads per lane.
// Packed layout per layer (halves): E0T 64x128 | E1T 64x64 | X0T 64x64 |
//                                   X1T 64x64 | X2T 16x64
// ---------------------------------------------------------------------------
__global__ void wprep_kernel(const float* __restrict__ pe_w0,
                             const float* __restrict__ pe_w1,
                             const float* __restrict__ px_w0,
                             const float* __restrict__ px_w1,
                             const float* __restrict__ px_w2,
                             f16* __restrict__ wPack) {
    int idx = blockIdx.x * blockDim.x + threadIdx.x;
    if (idx >= LL * WPL) return;
    int l = idx / WPL, r = idx % WPL;
    float v = 0.f;
    if (r < 8192) {                        // pe_w0 (100x64) -> [64][128] zero-padded
        int u = r >> 7, k = r & 127;
        v = (k < 100) ? pe_w0[(l * 100 + k) * UU + u] : 0.f;
    } else if (r < 12288) {                // pe_w1 (64x64)
        int r2 = r - 8192; int u = r2 >> 6, k = r2 & 63;
        v = pe_w1[(l * 64 + k) * UU + u];
    } else if (r < 16384) {                // px_w0
        int r2 = r - 12288; int u = r2 >> 6, k = r2 & 63;
        v = px_w0[(l * 64 + k) * UU + u];
    } else if (r < 20480) {                // px_w1
        int r2 = r - 16384; int u = r2 >> 6, k = r2 & 63;
        v = px_w1[(l * 64 + k) * UU + u];
    } else {                               // px_w2 (64x4) -> [16][64], rows >=4 zero
        int r2 = r - 20480; int u = r2 >> 6, k = r2 & 63;
        v = (u < HH) ? px_w2[(l * 64 + k) * HH + u] : 0.f;
    }
    wPack[idx] = (f16)v;
}

// ---------------------------------------------------------------------------
// Init: xh = broadcast(x) over heads; h = 1/N (mean_j softmax_j == 1/N exactly)
// ---------------------------------------------------------------------------
__global__ void init_kernel(const float* __restrict__ x,
                            float* __restrict__ xh, float* __restrict__ h) {
    int idx = blockIdx.x * blockDim.x + threadIdx.x;
    if (idx < NN * 12) {
        int i = idx / 12, d = idx % 3;
        xh[idx] = x[i * 3 + d];
    }
    if (idx < NN * DD) h[idx] = 1.f / (float)NN;
}

// ---------------------------------------------------------------------------
// Per-layer node features: hcat[j] = [h[j] (32), snh[j] (16)] in f16
// ---------------------------------------------------------------------------
__global__ void hcat_kernel(const float* __restrict__ xh,
                            const float* __restrict__ h,
                            f16* __restrict__ hcat) {
    int j = blockIdx.x * blockDim.x + threadIdx.x;
    if (j >= NN) return;
    float xl[12];
#pragma unroll
    for (int t = 0; t < 12; ++t) xl[t] = xh[j * 12 + t];
#pragma unroll
    for (int d = 0; d < DD; ++d) hcat[j * HC + d] = (f16)h[j * DD + d];
#pragma unroll
    for (int a = 0; a < HH; ++a)
#pragma unroll
        for (int b = 0; b < HH; ++b) {
            float s = 0.f;
#pragma unroll
            for (int d = 0; d < 3; ++d) {
                float dv = xl[a * 3 + d] - xl[b * 3 + d];
                s += dv * dv;
            }
            hcat[j * HC + DD + a * HH + b] = (f16)s;
        }
}

// ---------------------------------------------------------------------------
// Main EGCL pair kernel: one workgroup per node i, 4 waves sweep j-tiles of 16.
// ---------------------------------------------------------------------------
__global__ __launch_bounds__(128, 1)
void egcl_pair_kernel(const float* __restrict__ xhIn, const float* __restrict__ hIn,
                      const f16* __restrict__ hcat, const f16* __restrict__ wPack,
                      const float* __restrict__ pe_b0, const float* __restrict__ pe_b1,
                      const float* __restrict__ pi_w,  const float* __restrict__ pi_b,
                      const float* __restrict__ px_b0, const float* __restrict__ px_b1,
                      const float* __restrict__ px_b2,
                      const float* __restrict__ ph_w0, const float* __restrict__ ph_b0,
                      const float* __restrict__ ph_w1, const float* __restrict__ ph_b1,
                      const float* __restrict__ ph_w2, const float* __restrict__ ph_b2,
                      int layer, float* __restrict__ xhOut, float* __restrict__ hOut) {
    __shared__ __attribute__((aligned(16))) f16 stA[4][16 * KE];   // e_in / t0 / t1 staging
    __shared__ __attribute__((aligned(16))) f16 stM[4][16 * UU];   // m / t2 staging
    __shared__ __attribute__((aligned(16))) f16 stP[4][16 * 16];   // phi_x output
    __shared__ float eArr[4][16];
    __shared__ float mAcc[UU];
    __shared__ float sAcc[12];
    __shared__ float xiS[12];
    __shared__ __attribute__((aligned(16))) f16 hciS[HC];
    __shared__ float hiS[DD];
    __shared__ float piwS[UU];
    __shared__ float t0S[UU];
    __shared__ float t1S[UU];

    const int i   = blockIdx.x;
    const int tid = threadIdx.x;
    const int wv  = tid >> 5;
    const int ln  = tid & 31;
    const int lm  = ln & 15;
    const int hiL = ln >> 4;          // lane>=16 half-select

    const f16* wE0 = wPack + layer * WPL;
    const f16* wE1 = wE0 + 8192;
    const f16* wX0 = wE0 + 12288;
    const f16* wX1 = wE0 + 16384;
    const f16* wX2 = wE0 + 20480;

    if (tid < 12)                 xiS[tid]       = xhIn[i * 12 + tid];
    if (tid >= 16 && tid < 16+HC) hciS[tid - 16] = hcat[i * HC + (tid - 16)];
    if (tid >= 64 && tid < 96)    hiS[tid - 64]  = hIn[i * DD + (tid - 64)];
    if (tid < UU)                 mAcc[tid]      = 0.f;
    if (tid >= 96 && tid < 108)   sAcc[tid - 96] = 0.f;
    if (tid < UU)                 piwS[tid]      = pi_w[layer * UU + tid];
    __syncthreads();

    float xiL[12];
#pragma unroll
    for (int t = 0; t < 12; ++t) xiL[t] = xiS[t];

    float miA = 0.f, miB = 0.f;
    float shA[12];
#pragma unroll
    for (int t = 0; t < 12; ++t) shA[t] = 0.f;

    f16* SA = stA[wv];
    f16* SM = stM[wv];
    f16* SP = stP[wv];

    float xj[12], sq[HH];

    for (int tile = wv; tile < NN / 16; tile += 4) {
        const int j0 = tile * 16;

        // ---- stage e_in = [sqn(4) | hcat_j(48) | hcat_i(48) | pad] -----------
        if (ln < 16) {
            const int j = j0 + ln;
#pragma unroll
            for (int t = 0; t < 12; ++t) xj[t] = xhIn[j * 12 + t];
#pragma unroll
            for (int hh = 0; hh < HH; ++hh) {
                float s = 0.f;
#pragma unroll
                for (int d = 0; d < 3; ++d) {
                    float dv = xj[hh * 3 + d] - xiL[hh * 3 + d];
                    s += dv * dv;
                }
                sq[hh] = s;
                SA[ln * KE + hh] = (f16)s;
            }
            const f16* hj = hcat + j * HC;
#pragma unroll
            for (int t = 0; t < 12; ++t)
                *(h4*)(SA + ln * KE + 4 + t * 4) = *(const h4*)(hj + t * 4);
#pragma unroll
            for (int t = 0; t < 12; ++t)
                *(h4*)(SA + ln * KE + 52 + t * 4) = *(const h4*)(hciS + t * 4);
            h4 zz = (h4)(f16)0.f;
#pragma unroll
            for (int t = 0; t < 7; ++t)
                *(h4*)(SA + ln * KE + 100 + t * 4) = zz;
        }
        lds_fence();

        // ---- phi_e layer0: [16x128] x [128x64] -------------------------------
        v8f acc[4];
#pragma unroll
        for (int nt = 0; nt < 4; ++nt) acc[nt] = (v8f)0.f;
#pragma unroll
        for (int c = 0; c < 4; ++c) {
            v16h a = ld_frag(SA + lm * KE + c * 32 + hiL * 8);
#pragma unroll
            for (int nt = 0; nt < 4; ++nt) {
                v16h b = ld_frag(wE0 + (lm + nt * 16) * KE + c * 32 + hiL * 8);
                acc[nt] = __builtin_amdgcn_wmma_f32_16x16x32_f16(
                    false, a, false, b, (short)0, acc[nt], false, false);
            }
        }
#pragma unroll
        for (int nt = 0; nt < 4; ++nt) {
            float bb = pe_b0[layer * UU + lm + nt * 16];
#pragma unroll
            for (int r = 0; r < 8; ++r)
                SA[(r + 8 * hiL) * KE + lm + nt * 16] = (f16)silu_f(acc[nt][r] + bb);
        }
        lds_fence();

        // ---- phi_e layer1: [16x64] x [64x64] -> m ---------------------------
#pragma unroll
        for (int nt = 0; nt < 4; ++nt) acc[nt] = (v8f)0.f;
#pragma unroll
        for (int c = 0; c < 2; ++c) {
            v16h a = ld_frag(SA + lm * KE + c * 32 + hiL * 8);
#pragma unroll
            for (int nt = 0; nt < 4; ++nt) {
                v16h b = ld_frag(wE1 + (lm + nt * 16) * UU + c * 32 + hiL * 8);
                acc[nt] = __builtin_amdgcn_wmma_f32_16x16x32_f16(
                    false, a, false, b, (short)0, acc[nt], false, false);
            }
        }
#pragma unroll
        for (int nt = 0; nt < 4; ++nt) {
            float bb = pe_b1[layer * UU + lm + nt * 16];
#pragma unroll
            for (int r = 0; r < 8; ++r)
                SM[(r + 8 * hiL) * UU + lm + nt * 16] = (f16)silu_f(acc[nt][r] + bb);
        }
        lds_fence();

        // ---- phi_inf (e) + m_i accumulation ---------------------------------
        if (ln < 16) {
            float dot = pi_b[layer];
#pragma unroll
            for (int k = 0; k < UU; ++k) dot += (float)SM[ln * UU + k] * piwS[k];
            float e = 1.f / (1.f + __expf(-dot));
            if (j0 + ln == i) e = 0.f;   // diagonal masked
            eArr[wv][ln] = e;
        }
        lds_fence();
#pragma unroll
        for (int r = 0; r < 16; ++r) {
            float e = eArr[wv][r];
            miA += (float)SM[r * UU + ln] * e;
            miB += (float)SM[r * UU + ln + 32] * e;
        }

        // ---- phi_x layer0: m -> t1 (into SA) --------------------------------
#pragma unroll
        for (int nt = 0; nt < 4; ++nt) acc[nt] = (v8f)0.f;
#pragma unroll
        for (int c = 0; c < 2; ++c) {
            v16h a = ld_frag(SM + lm * UU + c * 32 + hiL * 8);
#pragma unroll
            for (int nt = 0; nt < 4; ++nt) {
                v16h b = ld_frag(wX0 + (lm + nt * 16) * UU + c * 32 + hiL * 8);
                acc[nt] = __builtin_amdgcn_wmma_f32_16x16x32_f16(
                    false, a, false, b, (short)0, acc[nt], false, false);
            }
        }
#pragma unroll
        for (int nt = 0; nt < 4; ++nt) {
            float bb = px_b0[layer * UU + lm + nt * 16];
#pragma unroll
            for (int r = 0; r < 8; ++r)
                SA[(r + 8 * hiL) * KE + lm + nt * 16] = (f16)silu_f(acc[nt][r] + bb);
        }
        lds_fence();

        // ---- phi_x layer1: t1 -> t2 (into SM) -------------------------------
#pragma unroll
        for (int nt = 0; nt < 4; ++nt) acc[nt] = (v8f)0.f;
#pragma unroll
        for (int c = 0; c < 2; ++c) {
            v16h a = ld_frag(SA + lm * KE + c * 32 + hiL * 8);
#pragma unroll
            for (int nt = 0; nt < 4; ++nt) {
                v16h b = ld_frag(wX1 + (lm + nt * 16) * UU + c * 32 + hiL * 8);
                acc[nt] = __builtin_amdgcn_wmma_f32_16x16x32_f16(
                    false, a, false, b, (short)0, acc[nt], false, false);
            }
        }
#pragma unroll
        for (int nt = 0; nt < 4; ++nt) {
            float bb = px_b1[layer * UU + lm + nt * 16];
#pragma unroll
            for (int r = 0; r < 8; ++r)
                SM[(r + 8 * hiL) * UU + lm + nt * 16] = (f16)silu_f(acc[nt][r] + bb);
        }
        lds_fence();

        // ---- phi_x layer2: t2 -> px [16x16], cols 0..3 valid ----------------
        {
            v8f accP = (v8f)0.f;
#pragma unroll
            for (int c = 0; c < 2; ++c) {
                v16h a = ld_frag(SM + lm * UU + c * 32 + hiL * 8);
                v16h b = ld_frag(wX2 + lm * UU + c * 32 + hiL * 8);
                accP = __builtin_amdgcn_wmma_f32_16x16x32_f16(
                    false, a, false, b, (short)0, accP, false, false);
            }
            float bb = (lm < HH) ? px_b2[layer * HH + lm] : 0.f;
#pragma unroll
            for (int r = 0; r < 8; ++r)
                SP[(r + 8 * hiL) * 16 + lm] = (f16)(accP[r] + bb);
        }
        lds_fence();

        // ---- equivariant shift accumulation ---------------------------------
        if (ln < 16) {
#pragma unroll
            for (int hh = 0; hh < HH; ++hh) {
                float pxv = (float)SP[ln * 16 + hh];
                float w = pxv / (sqrtf(sq[hh] + 1e-8f) + 1.f);
#pragma unroll
                for (int d = 0; d < 3; ++d)
                    shA[hh * 3 + d] += (xj[hh * 3 + d] - xiL[hh * 3 + d]) * w;
            }
        }
    }

    atomicAdd(&mAcc[ln], miA);
    atomicAdd(&mAcc[ln + 32], miB);
    if (ln < 16) {
#pragma unroll
        for (int t = 0; t < 12; ++t) atomicAdd(&sAcc[t], shA[t]);
    }
    __syncthreads();

    // ---- finalize: coords out + phi_h (tiny per-node MLP, wave 0 only) ------
    if (wv == 0) {
        if (ln < 12)
            xhOut[i * 12 + ln] = xiS[ln] + sAcc[ln] * (1.f / (float)(NN - 1));

        const float* W0 = ph_w0 + layer * 96 * UU;
        const float* W1 = ph_w1 + layer * UU * UU;
        const float* W2 = ph_w2 + layer * UU * DD;
#pragma unroll
        for (int half = 0; half < 2; ++half) {
            int u = ln + 32 * half;
            float a = ph_b0[layer * UU + u];
            for (int k = 0; k < UU; ++k) a += mAcc[k] * W0[k * UU + u];
            for (int k = 0; k < DD; ++k) a += hiS[k] * W0[(UU + k) * UU + u];
            t0S[u] = silu_f(a);
        }
        lds_fence();
#pragma unroll
        for (int half = 0; half < 2; ++half) {
            int u = ln + 32 * half;
            float a = ph_b1[layer * UU + u];
            for (int k = 0; k < UU; ++k) a += t0S[k] * W1[k * UU + u];
            t1S[u] = silu_f(a);
        }
        lds_fence();
        {
            float a = ph_b2[layer * DD + ln];
            for (int k = 0; k < UU; ++k) a += t1S[k] * W2[k * DD + ln];
            hOut[i * DD + ln] = hiS[ln] + a;
        }
    }
}

// ---------------------------------------------------------------------------
extern "C" void kernel_launch(void* const* d_in, const int* in_sizes, int n_in,
                              void* d_out, int out_size, void* d_ws, size_t ws_size,
                              hipStream_t stream) {
    const float* x     = (const float*)d_in[0];
    const float* pe_w0 = (const float*)d_in[3];
    const float* pe_b0 = (const float*)d_in[4];
    const float* pe_w1 = (const float*)d_in[5];
    const float* pe_b1 = (const float*)d_in[6];
    const float* pi_w  = (const float*)d_in[7];
    const float* pi_b  = (const float*)d_in[8];
    const float* px_w0 = (const float*)d_in[9];
    const float* px_b0 = (const float*)d_in[10];
    const float* px_w1 = (const float*)d_in[11];
    const float* px_b1 = (const float*)d_in[12];
    const float* px_w2 = (const float*)d_in[13];
    const float* px_b2 = (const float*)d_in[14];
    const float* ph_w0 = (const float*)d_in[15];
    const float* ph_b0 = (const float*)d_in[16];
    const float* ph_w1 = (const float*)d_in[17];
    const float* ph_b1 = (const float*)d_in[18];
    const float* ph_w2 = (const float*)d_in[19];
    const float* ph_b2 = (const float*)d_in[20];

    char* ws = (char*)d_ws;
    float* xhA  = (float*)(ws + 0);        // 512*12 f32
    float* xhB  = (float*)(ws + 24576);
    float* hA   = (float*)(ws + 49152);    // 512*32 f32
    float* hB   = (float*)(ws + 114688);
    f16*   hcat = (f16*)  (ws + 180224);   // 512*48 f16
    f16*  wPack = (f16*)  (ws + 229376);   // 3*21504 f16

    wprep_kernel<<<(LL * WPL + 255) / 256, 256, 0, stream>>>(
        pe_w0, pe_w1, px_w0, px_w1, px_w2, wPack);
    init_kernel<<<(NN * DD + 255) / 256, 256, 0, stream>>>(x, xhA, hA);

    float* xhIn = xhA;
    float* hIn  = hA;
    for (int l = 0; l < LL; ++l) {
        float* xhOut = (l == LL - 1) ? (float*)d_out : ((l & 1) ? xhA : xhB);
        float* hOut  = (l & 1) ? hA : hB;
        hcat_kernel<<<(NN + 127) / 128, 128, 0, stream>>>(xhIn, hIn, hcat);
        egcl_pair_kernel<<<NN, 128, 0, stream>>>(
            xhIn, hIn, hcat, wPack,
            pe_b0, pe_b1, pi_w, pi_b, px_b0, px_b1, px_b2,
            ph_w0, ph_b0, ph_w1, ph_b1, ph_w2, ph_b2,
            l, xhOut, hOut);
        xhIn = xhOut;
        hIn  = hOut;
    }
}
